// OptimizedNexusBlock_75969381532449
// MI455X (gfx1250) — matrix operand order repaired and checked by reading, not compiled, not run on hardware
//
#include <hip/hip_runtime.h>
#include <hip/hip_bf16.h>
#include <math.h>

// ---------------------------------------------------------------------------
// Problem constants (match reference)
// ---------------------------------------------------------------------------
#define NN   16384
#define DM   256
#define EE   262144
#define BB   8
#define TT   32
#define HH   8
#define HDD  32

typedef __attribute__((ext_vector_type(16))) __bf16 v16bf;
typedef __attribute__((ext_vector_type(8)))  float  v8f;

__device__ __forceinline__ float gelu_exact(float x) {
    return 0.5f * x * (1.0f + erff(x * 0.70710678118654752f));
}

// exactly one v_cvt_pk_bf16_f32: lo -> bits[15:0], hi -> bits[31:16]
__device__ __forceinline__ unsigned pk2(float lo, float hi) {
    unsigned r;
    asm("v_cvt_pk_bf16_f32 %0, %1, %2" : "=v"(r) : "v"(lo), "v"(hi));
    return r;
}

// ---------------------------------------------------------------------------
// Generic bf16-WMMA GEMM:  Y[M x Nout] = epi( X[M x K] @ W^T + bias )
// W stored row-major (Nout x K). K multiple of 32, Nout multiple of 128.
// EPI: 0 = bias, 1 = bias+gelu, 2 = residual: Y = res + gamma*(acc+bias)
//      (gamma == nullptr -> 1.0)
// Block: 256 threads (8 wave32). Block tile 64x128, k-step 32, double-buffered
// LDS, peeled software pipeline (branch-free interior, 1 barrier / k-step).
// Each wave: 16x64 strip = four v_wmma_f32_16x16x32_bf16 per k-step.
// LDS holds bf16; A as [m][k], B transposed as [n][k] so every fragment is two
// aligned 16B chunks (ds_load_b128). Chunk slots XOR-swizzled with (row>>2)&3.
// Global->bf16 conversion uses inline-asm v_cvt_pk_bf16_f32 so staging costs
// exactly 12 VALU ops per thread per k-step (no layout-fixup bit ops).
// NOTE: A-tile global loads are unguarded; for M=264 launches they read <60KB
// past the source buffer, which is still inside d_ws (results discarded by the
// guarded store).
// ---------------------------------------------------------------------------
template <int EPI>
__global__ __launch_bounds__(256)
void gemm_bf16_wmma(const float* __restrict__ X, int M, int K,
                    const float* __restrict__ W, const float* __restrict__ bias,
                    int Nout, float* Y, const float* res,
                    const float* __restrict__ gamma)
{
    __shared__ alignas(16) __bf16 lA[2][64][32];    // [pg][m][k]
    __shared__ alignas(16) __bf16 lB[2][128][32];   // [pg][n][k] (transposed)

    const int tid  = threadIdx.x;
    const int lane = tid & 31;
    const int wave = tid >> 5;
    const int mbase = blockIdx.x * 64;
    const int nbase = blockIdx.y * 128;
    const int mw = (wave & 3) * 16;   // wave row sub-tile
    const int nw = (wave >> 2) * 64;  // wave col sub-tile

    v8f acc0 = {}, acc1 = {}, acc2 = {}, acc3 = {};

    // cooperative staging mappings
    const int ar = tid >> 2;          // A: row 0..63
    const int ac = (tid & 3) * 8;     // A: k chunk {0,8,16,24}
    const int bn = tid >> 1;          // B: row (=n) 0..127
    const int bk = (tid & 1) * 16;    // B: k offset {0,16}

    const float* aptr = X + (size_t)(mbase + ar) * K + ac;
    const float* bptr = W + (size_t)(nbase + bn) * K + bk;

    auto pack8 = [](float4 a, float4 b) {
        uint4 u;
        u.x = pk2(a.x, a.y); u.y = pk2(a.z, a.w);
        u.z = pk2(b.x, b.y); u.w = pk2(b.z, b.w);
        return u;
    };
    // swizzled element offset of 8-element chunk `c` in row `row`
    auto sw = [](int row, int c) { return (c ^ ((row >> 2) & 3)) * 8; };

    // packed prefetch registers (12 VGPRs total)
    uint4 pa, pb0, pb1;
    auto loadpack = [&](int k0) {
        const float4 a0 = *(const float4*)(aptr + k0);
        const float4 a1 = *(const float4*)(aptr + k0 + 4);
        const float4 b0 = *(const float4*)(bptr + k0);
        const float4 b1 = *(const float4*)(bptr + k0 + 4);
        const float4 b2 = *(const float4*)(bptr + k0 + 8);
        const float4 b3 = *(const float4*)(bptr + k0 + 12);
        pa  = pack8(a0, a1);
        pb0 = pack8(b0, b1);
        pb1 = pack8(b2, b3);
    };
    const int aSt  = sw(ar, ac >> 3);
    const int bSt0 = sw(bn, (bk >> 3) + 0);
    const int bSt1 = sw(bn, (bk >> 3) + 1);
    auto store = [&](int pg) {
        *(uint4*)&lA[pg][ar][aSt]  = pa;
        *(uint4*)&lB[pg][bn][bSt0] = pb0;
        *(uint4*)&lB[pg][bn][bSt1] = pb1;
    };

    // fragment geometry (ISA 7.12.2), all offsets loop-invariant
    union FR { uint4 u[2]; v16bf v; };
    const int mrow = mw + (lane & 15);
    const int kga  = (lane >> 4) * 2;   // A chunk base: 0 or 2
    const int kgb  = (lane >> 4) * 2;   // B chunk base: 0 or 2
    const int nc   = lane & 15;
    const int aF0 = sw(mrow, kga + 0);
    const int aF1 = sw(mrow, kga + 1);
    const int n0 = nw + nc,      n1 = nw + 16 + nc;
    const int n2 = nw + 32 + nc, n3 = nw + 48 + nc;
    const int b00 = sw(n0, kgb), b01 = sw(n0, kgb + 1);
    const int b10 = sw(n1, kgb), b11 = sw(n1, kgb + 1);
    const int b20 = sw(n2, kgb), b21 = sw(n2, kgb + 1);
    const int b30 = sw(n3, kgb), b31 = sw(n3, kgb + 1);

    auto compute = [&](int pg) {
        FR fa;
        fa.u[0] = *(const uint4*)&lA[pg][mrow][aF0];
        fa.u[1] = *(const uint4*)&lA[pg][mrow][aF1];
        FR fb;
        fb.u[0] = *(const uint4*)&lB[pg][n0][b00];
        fb.u[1] = *(const uint4*)&lB[pg][n0][b01];
        acc0 = __builtin_amdgcn_wmma_f32_16x16x32_bf16(false, fa.v, false, fb.v,
                                                       (short)0, acc0, false, false);
        fb.u[0] = *(const uint4*)&lB[pg][n1][b10];
        fb.u[1] = *(const uint4*)&lB[pg][n1][b11];
        acc1 = __builtin_amdgcn_wmma_f32_16x16x32_bf16(false, fa.v, false, fb.v,
                                                       (short)0, acc1, false, false);
        fb.u[0] = *(const uint4*)&lB[pg][n2][b20];
        fb.u[1] = *(const uint4*)&lB[pg][n2][b21];
        acc2 = __builtin_amdgcn_wmma_f32_16x16x32_bf16(false, fa.v, false, fb.v,
                                                       (short)0, acc2, false, false);
        fb.u[0] = *(const uint4*)&lB[pg][n3][b30];
        fb.u[1] = *(const uint4*)&lB[pg][n3][b31];
        acc3 = __builtin_amdgcn_wmma_f32_16x16x32_bf16(false, fa.v, false, fb.v,
                                                       (short)0, acc3, false, false);
    };

    // peeled software pipeline
    loadpack(0);
    store(0);
    __syncthreads();
    int pg = 0;
    for (int k0 = 32; k0 < K; k0 += 32) {
        loadpack(k0);
        compute(pg);
        store(pg ^ 1);
        __syncthreads();
        pg ^= 1;
    }
    compute(pg);

    // epilogue: C/D layout: VGPR r -> row mw + (l>>4)*8 + r, col (l&15)
    const int mrow0 = mbase + mw + (lane >> 4) * 8;
    const int ncol0 = nbase + nw + nc;
    auto write_tile = [&](const v8f& a, int n) {
#pragma unroll
        for (int r = 0; r < 8; ++r) {
            const int m = mrow0 + r;
            if (m < M) {
                float v = a[r] + bias[n];
                if (EPI == 1) v = gelu_exact(v);
                if (EPI == 2) {
                    const float gmm = gamma ? gamma[n] : 1.0f;
                    v = res[(size_t)m * Nout + n] + gmm * v;
                }
                Y[(size_t)m * Nout + n] = v;
            }
        }
    };
    write_tile(acc0, ncol0);
    write_tile(acc1, ncol0 + 16);
    write_tile(acc2, ncol0 + 32);
    write_tile(acc3, ncol0 + 48);
}

// ---------------------------------------------------------------------------
// Row LayerNorm (optionally fused exact GELU). One block (256 thr) per row.
// ---------------------------------------------------------------------------
__global__ __launch_bounds__(256)
void ln_kernel(const float* X, int W, const float* __restrict__ g,
               const float* __restrict__ b, float* Y, int do_gelu)
{
    const int row = blockIdx.x;
    const float* x = X + (size_t)row * W;
    float* y = Y + (size_t)row * W;
    __shared__ float red[256];

    float s = 0.f, s2 = 0.f;
    for (int c = threadIdx.x; c < W; c += 256) {
        const float v = x[c];
        s += v; s2 += v * v;
    }
    red[threadIdx.x] = s; __syncthreads();
    for (int st = 128; st > 0; st >>= 1) {
        if (threadIdx.x < st) red[threadIdx.x] += red[threadIdx.x + st];
        __syncthreads();
    }
    const float mean = red[0] / (float)W;
    __syncthreads();
    red[threadIdx.x] = s2; __syncthreads();
    for (int st = 128; st > 0; st >>= 1) {
        if (threadIdx.x < st) red[threadIdx.x] += red[threadIdx.x + st];
        __syncthreads();
    }
    const float var  = red[0] / (float)W - mean * mean;
    const float rstd = rsqrtf(var + 1e-5f);
    for (int c = threadIdx.x; c < W; c += 256) {
        float v = (x[c] - mean) * rstd * g[c] + b[c];
        if (do_gelu) v = gelu_exact(v);
        y[c] = v;
    }
}

// ---------------------------------------------------------------------------
// hin = (1 + eps) * xn
// ---------------------------------------------------------------------------
__global__ __launch_bounds__(256)
void init_hin_kernel(const float* __restrict__ xn, const float* __restrict__ epsp,
                     float* __restrict__ hin, int total4)
{
    const int i = blockIdx.x * blockDim.x + threadIdx.x;
    if (i >= total4) return;
    const float e = 1.0f + epsp[0];
    float4 v = ((const float4*)xn)[i];
    v.x *= e; v.y *= e; v.z *= e; v.w *= e;
    ((float4*)hin)[i] = v;
}

// ---------------------------------------------------------------------------
// GINE message + scatter: hin[dst] += relu(xn[src] + edge_attr)
// ---------------------------------------------------------------------------
__global__ __launch_bounds__(256)
void edge_scatter_kernel(const float* __restrict__ xn, const float* __restrict__ ea,
                         const int* __restrict__ ei, float* hin)
{
    const int tid = blockIdx.x * blockDim.x + threadIdx.x;
    const int e   = tid >> 6;
    const int c4  = tid & 63;
    if (e >= EE) return;
    const int src = ei[e];
    const int dst = ei[EE + e];
    const float4 a = ((const float4*)(xn + (size_t)src * DM))[c4];
    const float4 w = ((const float4*)(ea + (size_t)e * DM))[c4];
    float4 m;
    m.x = fmaxf(a.x + w.x, 0.f); m.y = fmaxf(a.y + w.y, 0.f);
    m.z = fmaxf(a.z + w.z, 0.f); m.w = fmaxf(a.w + w.w, 0.f);
    float* p = hin + (size_t)dst * DM + c4 * 4;
    atomicAdd(p + 0, m.x); atomicAdd(p + 1, m.y);
    atomicAdd(p + 2, m.z); atomicAdd(p + 3, m.w);
}

__global__ void zero_kernel(float* p, int n)
{
    const int i = blockIdx.x * blockDim.x + threadIdx.x;
    if (i < n) p[i] = 0.f;
}

__global__ void count_kernel(const int* __restrict__ batch, float* cnt, int n)
{
    const int i = blockIdx.x * blockDim.x + threadIdx.x;
    if (i < n) atomicAdd(&cnt[batch[i]], 1.0f);
}

__global__ __launch_bounds__(256)
void sum_acc_kernel(const float* __restrict__ x1, const int* __restrict__ batch,
                    float* sums)
{
    const int tid = blockIdx.x * blockDim.x + threadIdx.x;
    const int i   = tid >> 6;
    const int c4  = tid & 63;
    if (i >= NN) return;
    const int b = batch[i];
    const float4 v = ((const float4*)(x1 + (size_t)i * DM))[c4];
    float* p = sums + b * DM + c4 * 4;
    atomicAdd(p + 0, v.x); atomicAdd(p + 1, v.y);
    atomicAdd(p + 2, v.z); atomicAdd(p + 3, v.w);
}

__global__ void sum_div_kernel(float* sums, const float* __restrict__ cnt)
{
    const int i = blockIdx.x * blockDim.x + threadIdx.x;
    if (i >= BB * DM) return;
    sums[i] /= fmaxf(cnt[i / DM], 1.0f);
}

// ctx (B, T+1, D) = concat(vn, node_summary)
__global__ void ctx_kernel(const float* __restrict__ vn, const float* __restrict__ sums,
                           float* __restrict__ ctx)
{
    const int tid = blockIdx.x * blockDim.x + threadIdx.x;
    if (tid >= BB * (TT + 1) * DM) return;
    const int r = tid / DM, c = tid % DM;
    const int b = r / (TT + 1), t = r % (TT + 1);
    ctx[tid] = (t < TT) ? vn[((size_t)b * TT + t) * DM + c] : sums[b * DM + c];
}

// ---------------------------------------------------------------------------
// Token MHA (B,T queries over T+1 keys), one thread per (b,h,q), online softmax
// ---------------------------------------------------------------------------
__global__ __launch_bounds__(256)
void mha_kernel(const float* __restrict__ qh, const float* __restrict__ kvh,
                float* __restrict__ out)
{
    const int tid = blockIdx.x * blockDim.x + threadIdx.x; // 0..2047
    const int b  = tid >> 8;
    const int h  = (tid >> 5) & 7;
    const int qi = tid & 31;
    const float* q = qh + ((size_t)(b * TT + qi)) * DM + h * HDD;
    float qr[HDD], acc[HDD];
#pragma unroll
    for (int d = 0; d < HDD; ++d) { qr[d] = q[d]; acc[d] = 0.f; }
    float m = -1e30f, ssum = 0.f;
    const float scale = 0.17677669529663687f; // 1/sqrt(32)
    for (int kt = 0; kt < TT + 1; ++kt) {
        const float* kr = kvh + ((size_t)(b * (TT + 1) + kt)) * (2 * DM) + h * HDD;
        float s = 0.f;
#pragma unroll
        for (int d = 0; d < HDD; ++d) s += qr[d] * kr[d];
        s *= scale;
        const float mn   = fmaxf(m, s);
        const float corr = __expf(m - mn);
        const float p    = __expf(s - mn);
        ssum = ssum * corr + p;
        const float* vr = kr + DM;
#pragma unroll
        for (int d = 0; d < HDD; ++d) acc[d] = acc[d] * corr + p * vr[d];
        m = mn;
    }
    const float inv = 1.0f / ssum;
    float* o = out + ((size_t)(b * TT + qi)) * DM + h * HDD;
#pragma unroll
    for (int d = 0; d < HDD; ++d) o[d] = acc[d] * inv;
}

// ---------------------------------------------------------------------------
// Node->token cross attention. One block per node, one wave32 per head.
// ---------------------------------------------------------------------------
__global__ __launch_bounds__(256)
void cross_attn_kernel(const float* __restrict__ qnode,
                       const float* __restrict__ kvnode,
                       const int* __restrict__ batch,
                       float* __restrict__ out)
{
    const int n = blockIdx.x;
    const int h = threadIdx.x >> 5;
    const int l = threadIdx.x & 31;
    const int b = batch[n];
    const float scale = 0.17677669529663687f;

    const float qv = qnode[(size_t)n * DM + h * HDD + l];   // lane l = dim d

    // lane l = key t : score
    float s = 0.f;
    const float* krow = kvnode + ((size_t)(b * TT + l)) * (2 * DM) + h * HDD;
#pragma unroll
    for (int d = 0; d < HDD; ++d) s += __shfl(qv, d, 32) * krow[d];
    s *= scale;

    float mx = s;
#pragma unroll
    for (int off = 16; off > 0; off >>= 1) mx = fmaxf(mx, __shfl_xor(mx, off, 32));
    const float p = __expf(s - mx);
    float sum = p;
#pragma unroll
    for (int off = 16; off > 0; off >>= 1) sum += __shfl_xor(sum, off, 32);
    const float attn = p / sum;

    // lane l = dim d : output
    float o = 0.f;
#pragma unroll
    for (int t = 0; t < TT; ++t) {
        const float a = __shfl(attn, t, 32);
        o += a * kvnode[((size_t)(b * TT + t)) * (2 * DM) + DM + h * HDD + l];
    }
    out[(size_t)n * DM + h * HDD + l] = o;
}

// ---------------------------------------------------------------------------
// Host-side orchestration
// ---------------------------------------------------------------------------
extern "C" void kernel_launch(void* const* d_in, const int* in_sizes, int n_in,
                              void* d_out, int out_size, void* d_ws, size_t ws_size,
                              hipStream_t stream)
{
    (void)in_sizes; (void)n_in; (void)out_size; (void)ws_size;

    const float* x       = (const float*)d_in[0];
    const int*   ei      = (const int*)  d_in[1];
    const float* ea      = (const float*)d_in[2];
    const float* v       = (const float*)d_in[3];
    const int*   batch   = (const int*)  d_in[4];
    const float* n1_g    = (const float*)d_in[5];
    const float* n1_b    = (const float*)d_in[6];
    const float* vn_g    = (const float*)d_in[7];
    const float* vn_b    = (const float*)d_in[8];
    const float* n3_g    = (const float*)d_in[9];
    const float* n3_b    = (const float*)d_in[10];
    const float* n4_g    = (const float*)d_in[11];
    const float* n4_b    = (const float*)d_in[12];
    const float* geps    = (const float*)d_in[13];
    const float* g_w1    = (const float*)d_in[14];
    const float* g_b1    = (const float*)d_in[15];
    const float* g_lng   = (const float*)d_in[16];
    const float* g_lnb   = (const float*)d_in[17];
    const float* g_w2    = (const float*)d_in[18];
    const float* g_b2    = (const float*)d_in[19];
    const float* gamma1  = (const float*)d_in[20];
    const float* gamma2  = (const float*)d_in[21];
    const float* gamma3  = (const float*)d_in[22];
    const float* mha_in_w  = (const float*)d_in[23];
    const float* mha_in_b  = (const float*)d_in[24];
    const float* mha_out_w = (const float*)d_in[25];
    const float* mha_out_b = (const float*)d_in[26];
    const float* q_w     = (const float*)d_in[27];
    const float* q_b     = (const float*)d_in[28];
    const float* kv_w    = (const float*)d_in[29];
    const float* kv_b    = (const float*)d_in[30];
    const float* o_w     = (const float*)d_in[31];
    const float* o_b     = (const float*)d_in[32];
    const float* f_w1    = (const float*)d_in[33];
    const float* f_b1    = (const float*)d_in[34];
    const float* f_w2    = (const float*)d_in[35];
    const float* f_b2    = (const float*)d_in[36];

    float* out = (float*)d_out;
    float* ws  = (float*)d_ws;

    const size_t ND = (size_t)NN * DM;
    float* bufA   = ws;                 // xn / xn3 / attn_out / xn4  (N*D)
    float* bufB   = bufA + ND;          // hin / qnode                (N*D)
    float* bufC   = bufB + ND;          // t2d early, t4d late        (N*4D)
    float* bufD   = bufC + 4 * ND;      // x1 / x2                    (N*D)
    float* sums   = bufD + ND;          // B*D
    float* cnt    = sums + BB * DM;     // B
    float* vnb    = cnt + BB;           // B*T*D
    float* ctx    = vnb + BB * TT * DM;           // B*(T+1)*D
    float* qh     = ctx + BB * (TT + 1) * DM;     // B*T*D
    float* kvh    = qh + BB * TT * DM;            // B*(T+1)*2D
    float* mhao   = kvh + BB * (TT + 1) * 2 * DM; // B*T*D
    float* kvnode = mhao + BB * TT * DM;          // B*T*2D
    float* v_new  = out + ND;                     // (B,T,D) output region

    // ---- Stage 1: GINE + MLP + residual -----------------------------------
    ln_kernel<<<NN, 256, 0, stream>>>(x, DM, n1_g, n1_b, bufA, 0);           // xn
    init_hin_kernel<<<(int)(ND / 4 + 255) / 256, 256, 0, stream>>>(bufA, geps, bufB, (int)(ND / 4));
    edge_scatter_kernel<<<(EE * 64) / 256, 256, 0, stream>>>(bufA, ea, ei, bufB); // hin
    gemm_bf16_wmma<0><<<dim3(NN / 64, 512 / 128), 256, 0, stream>>>(
        bufB, NN, DM, g_w1, g_b1, 512, bufC, nullptr, nullptr);
    ln_kernel<<<NN, 256, 0, stream>>>(bufC, 512, g_lng, g_lnb, bufC, 1);     // LN+gelu
    gemm_bf16_wmma<2><<<dim3(NN / 64, DM / 128), 256, 0, stream>>>(
        bufC, NN, 512, g_w2, g_b2, DM, bufD, x, gamma1);                     // x1

    // ---- Stage 2: token MHA ----------------------------------------------
    zero_kernel<<<(BB * DM + BB + 255) / 256, 256, 0, stream>>>(sums, BB * DM + BB);
    sum_acc_kernel<<<(NN * 64) / 256, 256, 0, stream>>>(bufD, batch, sums);
    count_kernel<<<(NN + 255) / 256, 256, 0, stream>>>(batch, cnt, NN);
    sum_div_kernel<<<(BB * DM + 255) / 256, 256, 0, stream>>>(sums, cnt);
    ln_kernel<<<BB * TT, 256, 0, stream>>>(v, DM, vn_g, vn_b, vnb, 0);       // vn
    ctx_kernel<<<(BB * (TT + 1) * DM + 255) / 256, 256, 0, stream>>>(vnb, sums, ctx);
    gemm_bf16_wmma<0><<<dim3((BB * TT + 63) / 64, DM / 128), 256, 0, stream>>>(
        vnb, BB * TT, DM, mha_in_w, mha_in_b, DM, qh, nullptr, nullptr);
    gemm_bf16_wmma<0><<<dim3((BB * (TT + 1) + 63) / 64, 512 / 128), 256, 0, stream>>>(
        ctx, BB * (TT + 1), DM, mha_in_w + (size_t)DM * DM, mha_in_b + DM,
        512, kvh, nullptr, nullptr);
    mha_kernel<<<(BB * HH * TT) / 256, 256, 0, stream>>>(qh, kvh, mhao);
    gemm_bf16_wmma<2><<<dim3((BB * TT + 63) / 64, DM / 128), 256, 0, stream>>>(
        mhao, BB * TT, DM, mha_out_w, mha_out_b, DM, v_new, v, gamma2);      // v out

    // ---- Stage 3: node->token cross attention ----------------------------
    ln_kernel<<<NN, 256, 0, stream>>>(bufD, DM, n3_g, n3_b, bufA, 0);        // xn3
    gemm_bf16_wmma<0><<<dim3(NN / 64, DM / 128), 256, 0, stream>>>(
        bufA, NN, DM, q_w, q_b, DM, bufB, nullptr, nullptr);                 // qnode
    gemm_bf16_wmma<0><<<dim3((BB * TT + 63) / 64, 512 / 128), 256, 0, stream>>>(
        v_new, BB * TT, DM, kv_w, kv_b, 512, kvnode, nullptr, nullptr);
    cross_attn_kernel<<<NN, 256, 0, stream>>>(bufB, kvnode, batch, bufA);    // attn_out
    gemm_bf16_wmma<2><<<dim3(NN / 64, DM / 128), 256, 0, stream>>>(
        bufA, NN, DM, o_w, o_b, DM, bufD, bufD, gamma3);                     // x2 (in place)

    // ---- Stage 4: FFN -----------------------------------------------------
    ln_kernel<<<NN, 256, 0, stream>>>(bufD, DM, n4_g, n4_b, bufA, 0);        // xn4
    gemm_bf16_wmma<1><<<dim3(NN / 64, 1024 / 128), 256, 0, stream>>>(
        bufA, NN, DM, f_w1, f_b1, 1024, bufC, nullptr, nullptr);             // gelu
    gemm_bf16_wmma<2><<<dim3(NN / 64, DM / 128), 256, 0, stream>>>(
        bufC, NN, 1024, f_w2, f_b2, DM, out, bufD, nullptr);                 // x final
}
